// QuantLinear_90572270338513
// MI455X (gfx1250) — matrix operand order repaired and checked by reading, not compiled, not run on hardware
//
#include <hip/hip_runtime.h>

// Problem dims (fixed by the reference harness)
#define M_DIM   8192      // B*S = 4*2048
#define IN_DIM  4096      // K
#define OUT_DIM 16384     // N
#define QBLOCK  64

#define BK   32           // K step per WMMA (bf16 16x16x32)
#define LDA  40           // LDS row stride in elements (32 + 8 pad, keeps 16B align)
#define TILE_ELEMS (128 * LDA)       // 5120 elements per tile buffer
#define TILE_BYTES (TILE_ELEMS * 2)  // 10240 B per tile buffer

typedef __attribute__((ext_vector_type(16))) __bf16 v16bf;
typedef __attribute__((ext_vector_type(8)))  __bf16 v8bf;
typedef __attribute__((ext_vector_type(8)))  float  v8f;
typedef __attribute__((ext_vector_type(4)))  int      v4i;
typedef __attribute__((ext_vector_type(4)))  float    v4f;
typedef __attribute__((ext_vector_type(4)))  unsigned v4u;
typedef __attribute__((ext_vector_type(8)))  int      i32x8;
typedef __attribute__((ext_vector_type(4)))  int      i32x4;
typedef __attribute__((ext_vector_type(4)))  unsigned u32x4;

__device__ __constant__ float NF4[16] = {
    -1.0f, -0.6961928009986877f, -0.5250730514526367f, -0.39491748809814453f,
    -0.28444138169288635f, -0.18477343022823334f, -0.09105003625154495f, 0.0f,
    0.07958029955625534f, 0.16093020141124725f, 0.24611230194568634f, 0.33791524171829224f,
    0.44070982933044434f, 0.5626170039176941f, 0.7229568362236023f, 1.0f
};

// float -> bf16 bits, round-to-nearest-even
__device__ __forceinline__ unsigned short f2bf(float f) {
    unsigned int u = __float_as_uint(f);
    unsigned int r = u + 0x7FFFu + ((u >> 16) & 1u);
    return (unsigned short)(r >> 16);
}

// ---------------------------------------------------------------------------
// TDM: 2D tile load, 128 rows x 32 bf16, row stride 4096 elems in memory,
// deposited into LDS with 16B pad after each 64B row (=> LDA=40 elem stride).
// Descriptor packing per CDNA5 ISA.
// ---------------------------------------------------------------------------
__device__ __forceinline__ void tdm_load_tile_a(unsigned lds_off,
                                                unsigned long long gaddr) {
    u32x4 g0;
    g0.x = 0x1u;                                    // count=1, user mode, no gather
    g0.y = lds_off;                                 // lds_addr [63:32]
    g0.z = (unsigned)gaddr;                         // global_addr [95:64]
    g0.w = ((unsigned)(gaddr >> 32) & 0x01FFFFFFu)  // global_addr [120:96]
         | 0x80000000u;                             // type=2 ("image") [127:126]
    i32x8 g1;
    g1[0] = (1 << 16)        // data_size = 2 bytes
          | (1 << 20)        // pad_enable
          | (3 << 22)        // pad_interval: 16 DWORDs (64B) between pads
          | (3 << 25);       // pad_amount: 4 DWORDs (16B)
    g1[1] = (int)(4096u << 16);  // tensor_dim0 = 4096 (lo16 at bits 63:48)
    g1[2] = (int)(128u << 16);   // tensor_dim0 hi=0 ; tensor_dim1 = 128 (lo16)
    g1[3] = (int)(32u << 16);    // tensor_dim1 hi=0 ; tile_dim0 = 32
    g1[4] = 128;                 // tile_dim1 = 128 ; tile_dim2 = 0
    g1[5] = 4096;                // tensor_dim0_stride lo32 = 4096
    g1[6] = 0;                   // stride hi ; tensor_dim1_stride lo
    g1[7] = 0;                   // tensor_dim1_stride hi
    i32x4 g2 = {1, 1, 0, 0};         // tensor_dim2=1, tensor_dim3=1, stride2=0, tile_dim3=0
    i32x4 g3 = {0, (int)(1u << 16), 0, 0};  // stride3=0, tensor_dim4=1, tile_dim4=0
#if defined(__clang_major__) && (__clang_major__ >= 23)
    i32x8 g2w = {g2.x, g2.y, g2.z, g2.w, 0, 0, 0, 0};
    __builtin_amdgcn_tensor_load_to_lds(g0, g1, g2, g3, g2w, 0);
#else
    __builtin_amdgcn_tensor_load_to_lds(g0, g1, g2, g3, 0);
#endif
}

// ---------------------------------------------------------------------------
// Kernel 1: NF4 blockwise dequant -> bf16 W [OUT, IN]
// Codebook lookup via ds_bpermute (lanes 0..15 hold NF4 table), 8 elems/thread.
// ---------------------------------------------------------------------------
__global__ __launch_bounds__(256) void nf4_dequant_kernel(
    const int* __restrict__ codes, const float* __restrict__ absmax,
    unsigned short* __restrict__ wbf) {
    const long long t8   = (long long)blockIdx.x * 256 + threadIdx.x;
    const long long base = t8 * 8;                 // 8 consecutive elems along IN

    // lane-resident codebook: lane i (i<16) holds NF4[i]; repeated in lanes 16..31
    const unsigned cb = __float_as_uint(NF4[threadIdx.x & 15]);

    const int o  = (int)(base >> 12);              // / IN_DIM
    const int ib = (int)(base & (IN_DIM - 1));
    const float scale = absmax[(size_t)o * (IN_DIM / QBLOCK) + (ib >> 6)];

    const v4i c0 = __builtin_nontemporal_load(((const v4i*)codes) + t8 * 2);
    const v4i c1 = __builtin_nontemporal_load(((const v4i*)codes) + t8 * 2 + 1);
    int cs[8] = {c0.x, c0.y, c0.z, c0.w, c1.x, c1.y, c1.z, c1.w};

    v4u packed;
#pragma unroll
    for (int i = 0; i < 4; ++i) {
        float w0 = __uint_as_float(__builtin_amdgcn_ds_bpermute(cs[2*i]     << 2, (int)cb)) * scale;
        float w1 = __uint_as_float(__builtin_amdgcn_ds_bpermute(cs[2*i + 1] << 2, (int)cb)) * scale;
        packed[i] = (unsigned)f2bf(w0) | ((unsigned)f2bf(w1) << 16);
    }
    ((v4u*)wbf)[t8] = packed;
}

// ---------------------------------------------------------------------------
// Kernel 2: x f32 -> bf16, 8 elems/thread (read-once stream -> NT loads)
// ---------------------------------------------------------------------------
__global__ __launch_bounds__(256) void f32_to_bf16_kernel(
    const float* __restrict__ x, unsigned short* __restrict__ xb) {
    const long long t = (long long)blockIdx.x * 256 + threadIdx.x;
    const v4f a = __builtin_nontemporal_load(((const v4f*)x) + 2 * t);
    const v4f b = __builtin_nontemporal_load(((const v4f*)x) + 2 * t + 1);
    v4u o;
    o.x = (unsigned)f2bf(a.x) | ((unsigned)f2bf(a.y) << 16);
    o.y = (unsigned)f2bf(a.z) | ((unsigned)f2bf(a.w) << 16);
    o.z = (unsigned)f2bf(b.x) | ((unsigned)f2bf(b.y) << 16);
    o.w = (unsigned)f2bf(b.z) | ((unsigned)f2bf(b.w) << 16);
    ((v4u*)xb)[t] = o;
}

// ---------------------------------------------------------------------------
// Kernel 3: C[M,N] = A[M,K] (bf16) @ W[N,K]^T (bf16), f32 accum via WMMA.
// 256 threads = 8 waves (wave32). Block tile 128x128, BK=32.
// Waves: 2 (M) x 4 (N); each wave: 4x2 tiles of 16x16 -> 8 wmma/K-step.
// A tile staged by the Tensor Data Mover (1 descriptor, wave 0, TENSORcnt);
// B tile staged by per-lane global_load_async_to_lds_b128 (ASYNCcnt).
// Triple-buffered LDS, two tiles in flight, one barrier per K-step.
// Buffer rotation maintained incrementally (no % 3 division chains).
// ---------------------------------------------------------------------------
__global__ __launch_bounds__(256) void gemm_bf16_wmma_kernel(
    const unsigned short* __restrict__ A,   // [M, K] bf16
    const unsigned short* __restrict__ Bw,  // [N, K] bf16 (W rows = B columns)
    float* __restrict__ C) {                // [M, N] f32
    __shared__ unsigned short lds_a[3][TILE_ELEMS];
    __shared__ unsigned short lds_b[3][TILE_ELEMS];

    const int tid  = threadIdx.x;
    const int m0   = blockIdx.y * 128;
    const int n0   = blockIdx.x * 128;
    const int wave = tid >> 5;
    const int lane = tid & 31;
    const int wm   = wave & 1;       // 0..1 -> 64-row M slab
    const int wn   = wave >> 1;      // 0..3 -> 32-col N slab
    const int l15  = lane & 15;
    const int lh   = lane >> 4;      // lane half selects K sub-chunk per ISA layout
    // uniform-per-wave scalar wave id -> plain scalar branch for TDM issue
    const int waveId = __builtin_amdgcn_readfirstlane(wave);

    // B staging: 512 16B chunks per tile, 2 per thread
    const unsigned short* gB[2];
    unsigned lbOff[2];
#pragma unroll
    for (int j = 0; j < 2; ++j) {
        int c   = tid + 256 * j;
        int row = c >> 2;            // 4 chunks of 8 elems per 32-elem row
        int kc  = (c & 3) * 8;
        gB[j] = Bw + (size_t)(n0 + row) * IN_DIM + kc;
        lbOff[j] = (unsigned)(uintptr_t)&lds_b[0][row * LDA + kc];
    }
    const unsigned laBase = (unsigned)(uintptr_t)&lds_a[0][0];
    // per-wave fragment base offsets (elements) within one tile buffer
    unsigned aFragOff[4], bFragOff[2];
#pragma unroll
    for (int i = 0; i < 4; ++i)
        aFragOff[i] = (unsigned)((wm * 64 + i * 16 + l15) * LDA + lh * 8);
#pragma unroll
    for (int j = 0; j < 2; ++j)
        bFragOff[j] = (unsigned)((wn * 32 + j * 16 + l15) * LDA + lh * 8);

    auto issue_b = [&](const unsigned short* gKB[2], unsigned boff) {
#pragma unroll
        for (int j = 0; j < 2; ++j) {
            unsigned long long gb = (unsigned long long)gKB[j];
            asm volatile("global_load_async_to_lds_b128 %0, %1, off"
                         :: "v"(lbOff[j] + boff), "v"(gb) : "memory");
        }
    };

    const v8f vzero = {0.f, 0.f, 0.f, 0.f, 0.f, 0.f, 0.f, 0.f};
    v8f acc[4][2];
#pragma unroll
    for (int i = 0; i < 4; ++i)
#pragma unroll
        for (int j = 0; j < 2; ++j) acc[i][j] = vzero;

    const int KSTEPS = IN_DIM / BK;  // 128

    // rotating state, maintained incrementally
    unsigned curOff = 0;                         // byte offset of tile being computed
    unsigned nxtOff = 2u * TILE_BYTES;           // byte offset for tile kk+2 refill
    unsigned long long gaNext =                  // TDM global addr of tile kk+2 (bytes)
        (unsigned long long)(A + (size_t)m0 * IN_DIM);
    const unsigned short* gBn[2] = {gB[0], gB[1]};  // async B addr of tile kk+2

    // prologue: two tiles in flight
    if (waveId == 0) {
        tdm_load_tile_a(laBase, gaNext);
        tdm_load_tile_a(laBase + TILE_BYTES, gaNext + (unsigned long long)BK * 2ull);
    }
    issue_b(gBn, 0);
#pragma unroll
    for (int j = 0; j < 2; ++j) gBn[j] += BK;
    issue_b(gBn, TILE_BYTES);
#pragma unroll
    for (int j = 0; j < 2; ++j) gBn[j] += BK;
    gaNext += (unsigned long long)(2 * BK) * 2ull;

    for (int kk = 0; kk < KSTEPS; ++kk) {
        // retire oldest tile (both TDM and async loads complete in order per wave)
        if (kk + 1 < KSTEPS) {
            if (waveId == 0) __builtin_amdgcn_s_wait_tensorcnt((short)1);
            asm volatile("s_wait_asynccnt 0x2" ::: "memory");
        } else {
            if (waveId == 0) __builtin_amdgcn_s_wait_tensorcnt((short)0);
            asm volatile("s_wait_asynccnt 0x0" ::: "memory");
        }
        __syncthreads();

        const unsigned curE = curOff >> 1;   // element offset of current tile

        // A fragments: 16x32 bf16; lane-half picks K {0..7,16..23} vs {8..15,24..31}
        v16bf af[4];
#pragma unroll
        for (int i = 0; i < 4; ++i) {
            const unsigned short* p = &lds_a[0][curE + aFragOff[i]];
            v8bf lo = *(const v8bf*)p;          // K = kb .. kb+7
            v8bf hi = *(const v8bf*)(p + 16);   // K = kb+16 .. kb+23
            af[i] = __builtin_shufflevector(lo, hi,
                0, 1, 2, 3, 4, 5, 6, 7, 8, 9, 10, 11, 12, 13, 14, 15);
        }
        // B fragments: 32x16 bf16; lane = column n, lane-half picks K {0..7,16..23} vs {8..15,24..31}
        v16bf bfr[2];
#pragma unroll
        for (int j = 0; j < 2; ++j) {
            const unsigned short* p = &lds_b[0][curE + bFragOff[j]];
            v8bf lo = *(const v8bf*)p;
            v8bf hi = *(const v8bf*)(p + 16);
            bfr[j] = __builtin_shufflevector(lo, hi,
                0, 1, 2, 3, 4, 5, 6, 7, 8, 9, 10, 11, 12, 13, 14, 15);
        }

#pragma unroll
        for (int i = 0; i < 4; ++i)
#pragma unroll
            for (int j = 0; j < 2; ++j)
                acc[i][j] = __builtin_amdgcn_wmma_f32_16x16x32_bf16(
                    false, af[i], false, bfr[j], (short)0, acc[i][j], false, false);

        // refill: tile kk+2 into the buffer read at iteration kk-1
        // (all waves passed this iteration's barrier, so that buffer is free)
        if (kk + 2 < KSTEPS) {
            if (waveId == 0) tdm_load_tile_a(laBase + nxtOff, gaNext);
            issue_b(gBn, nxtOff);
#pragma unroll
            for (int j = 0; j < 2; ++j) gBn[j] += BK;
            gaNext += (unsigned long long)BK * 2ull;
        }

        // rotate buffers incrementally (no division)
        curOff += TILE_BYTES; if (curOff == 3u * TILE_BYTES) curOff = 0;
        nxtOff += TILE_BYTES; if (nxtOff == 3u * TILE_BYTES) nxtOff = 0;
    }

    // Epilogue: C/D layout — VGPR v holds M = v + lh*8, N = l15.
    // Output is write-once, never re-read -> non-temporal stores keep W/X in L2.
#pragma unroll
    for (int i = 0; i < 4; ++i) {
        const int mrow = m0 + wm * 64 + i * 16 + lh * 8;
#pragma unroll
        for (int j = 0; j < 2; ++j) {
            const int ncol = n0 + wn * 32 + j * 16 + l15;
            float* cp = C + (size_t)mrow * OUT_DIM + ncol;
#pragma unroll
            for (int v = 0; v < 8; ++v)
                __builtin_nontemporal_store(acc[i][j][v], cp + (size_t)v * OUT_DIM);
        }
    }
}

// ---------------------------------------------------------------------------
extern "C" void kernel_launch(void* const* d_in, const int* in_sizes, int n_in,
                              void* d_out, int out_size, void* d_ws, size_t ws_size,
                              hipStream_t stream) {
    (void)in_sizes; (void)n_in; (void)out_size; (void)ws_size;
    const float* x       = (const float*)d_in[0];
    const int*   w_codes = (const int*)d_in[1];
    const float* absmax  = (const float*)d_in[2];
    float*       out     = (float*)d_out;

    // workspace: bf16 W [OUT,IN] then bf16 X [M,IN]  (~192 MiB)
    unsigned short* wW = (unsigned short*)d_ws;
    unsigned short* wX = wW + (size_t)OUT_DIM * IN_DIM;

    const long long nW8 = (long long)OUT_DIM * IN_DIM / 8;   // elems/8
    nf4_dequant_kernel<<<(int)(nW8 / 256), 256, 0, stream>>>(w_codes, absmax, wW);

    const long long nX8 = (long long)M_DIM * IN_DIM / 8;
    f32_to_bf16_kernel<<<(int)(nX8 / 256), 256, 0, stream>>>(x, wX);

    dim3 grid(OUT_DIM / 128, M_DIM / 128);   // (128, 64)
    gemm_bf16_wmma_kernel<<<grid, 256, 0, stream>>>(wX, wW, out);
}
